// GATNet_53970559042043
// MI455X (gfx1250) — matrix operand order repaired and checked by reading, not compiled, not run on hardware
//
#include <hip/hip_runtime.h>
#include <cmath>
#include <cstdint>

typedef __attribute__((ext_vector_type(16))) _Float16 v16h;
typedef __attribute__((ext_vector_type(8)))  float    v8f;
typedef __attribute__((ext_vector_type(4)))  unsigned int v4u;
typedef __attribute__((ext_vector_type(8)))  int      v8i;
typedef __attribute__((ext_vector_type(4)))  int      v4i;

#define NEG_SLOPE 0.2f
#define NGRAPHS 64

#if __has_builtin(__builtin_amdgcn_tensor_load_to_lds) && \
    __has_builtin(__builtin_amdgcn_s_wait_tensorcnt)
#define USE_TDM 1
#else
#define USE_TDM 0
#endif

static inline int ceil_div(long a, int b) { return (int)((a + b - 1) / b); }

// ---------------- generic fill ----------------
__global__ void k_fill_f32(float* p, long n, float v) {
  long i = (long)blockIdx.x * blockDim.x + threadIdx.x;
  if (i < n) p[i] = v;
}

// ---------------- pack f32 [R,C] -> f16 [Rp,Cp] zero-padded ----------------
__global__ void k_pack_h(const float* __restrict__ src, _Float16* __restrict__ dst,
                         int R, int C, int Rp, int Cp) {
  long i = (long)blockIdx.x * blockDim.x + threadIdx.x;
  long total = (long)Rp * Cp;
  if (i >= total) return;
  int r = (int)(i / Cp), c = (int)(i % Cp);
  float v = (r < R && c < C) ? src[(size_t)r * C + c] : 0.0f;
  dst[i] = (_Float16)v;
}

#if USE_TDM
// ---- Tensor Data Mover: 2D tile DMA global -> LDS (D# per CDNA5 ISA ch.8) ----
// tensor dims == tile dims (buffers pre-padded, so no OOB clipping wanted).
// amdgpu-toolchain (clang-23) 6-arg builtin form: (v4u, v8i, v4i, v4i, v8i, i32 cpol).
__device__ __forceinline__ void tdm_load_2d(const void* gaddr, unsigned lds_byte_off,
                                            unsigned tile_w, unsigned tile_h,
                                            unsigned stride_elems) {
  unsigned long long ga = (unsigned long long)(uintptr_t)gaddr;
  v4u g0;
  g0.x = 1u;                                              // count=1 (valid, user mode)
  g0.y = lds_byte_off;                                    // lds_addr
  g0.z = (unsigned)ga;                                    // global_addr[31:0]
  g0.w = (unsigned)((ga >> 32) & 0x01FFFFFFu) | (2u << 30); // addr[56:32] | type=2
  v8i g1;
  g1[0] = (int)(1u << 16);                                // data_size=1 (2 bytes), mask=0
  g1[1] = (int)((tile_w & 0xFFFFu) << 16);                // tensor_dim0[15:0] @ bits[63:48]
  g1[2] = (int)(((tile_w >> 16) & 0xFFFFu) | ((tile_h & 0xFFFFu) << 16)); // dim0 hi | dim1 lo
  g1[3] = (int)(((tile_h >> 16) & 0xFFFFu) | (tile_w << 16));             // dim1 hi | tile_dim0
  g1[4] = (int)(tile_h & 0xFFFFu);                        // tile_dim1 (tile_dim2=0 -> 2D)
  g1[5] = (int)stride_elems;                              // tensor_dim0_stride[31:0]
  g1[6] = 0;                                              // stride hi | dim1_stride lo
  g1[7] = 0;
  v4i g2 = {};
  v4i g3 = {};
  v8i g4 = {};
  __builtin_amdgcn_tensor_load_to_lds(g0, g1, g2, g3, g4, 0);
}
#endif

// ---------------- tiled WMMA f16 GEMM: C[Mp,Np] = A[Mp,Kp] * B[Kp,Np] ----------------
// Mp multiple of 128, Np multiple of 64, Kp multiple of 32. Row-major everywhere.
// TDM double-buffers A(128x32) and B(32x64) tiles in LDS; 8 waves do 32x32 macro-tiles.
// Dynamic LDS: A0 @0 (8KB), A1 @8192, B0 @16384 (4KB), B1 @20480 -> 24KB total.
#define GEMM_LDS_BYTES 24576
__global__ __launch_bounds__(256) void k_gemm_wmma(
    const _Float16* __restrict__ A, const _Float16* __restrict__ B,
    float* __restrict__ C, int Mp, int Np, int Kp) {
  extern __shared__ _Float16 smem[];

  const int t    = threadIdx.x;
  const int lane = t & 31;
  const int wave = t >> 5;
  const int wy   = wave >> 1;       // 0..3 -> 32-row strip
  const int wx   = wave & 1;        // 0..1 -> 32-col strip
  const int hi   = lane >> 4;
  const int lo   = lane & 15;

  const int m0 = blockIdx.y * 128;
  const int n0 = blockIdx.x * 64;

  v8f acc[2][2] = {};

#if USE_TDM
  if (wave == 0) {  // per-wave DMA engine ops; EXEC ignored by TDM
    tdm_load_2d(A + (size_t)m0 * Kp, 0u,      32u, 128u, (unsigned)Kp);
    tdm_load_2d(B + n0,              16384u,  64u, 32u,  (unsigned)Np);
  }
#endif

  int buf = 0;
  for (int kc = 0; kc < Kp; kc += 32) {
    const bool has_next = (kc + 32) < Kp;
    // double-buffer bases, computed arithmetically (no LDS pointer arrays)
    const int aBase = buf * 4096;            // halfs: A0 @0, A1 @4096
    const int bBase = 8192 + buf * 2048;     // halfs: B0 @8192, B1 @10240
#if USE_TDM
    if (wave == 0) {
      if (has_next) {
        const unsigned aoff = (buf ^ 1) ? 8192u : 0u;          // bytes
        const unsigned boff = (buf ^ 1) ? 20480u : 16384u;     // bytes
        tdm_load_2d(A + (size_t)m0 * Kp + (kc + 32), aoff, 32u, 128u, (unsigned)Kp);
        tdm_load_2d(B + (size_t)(kc + 32) * Np + n0, boff, 64u, 32u, (unsigned)Np);
        __builtin_amdgcn_s_wait_tensorcnt(2);  // in-order: current pair has landed
      } else {
        __builtin_amdgcn_s_wait_tensorcnt(0);
      }
    }
#else
    {   // manual staging fallback: A dense [128][32], B dense [32][64]
      const int ar = t >> 1, ak = (t & 1) << 4;
      const uint4* gA = reinterpret_cast<const uint4*>(A + (size_t)(m0 + ar) * Kp + kc + ak);
      uint4 av0 = gA[0], av1 = gA[1];
      uint4* spA = reinterpret_cast<uint4*>(&smem[aBase + ar * 32 + ak]);
      spA[0] = av0; spA[1] = av1;
      const int bk = t >> 3, bn = (t & 7) << 3;
      uint4 bv = *reinterpret_cast<const uint4*>(B + (size_t)(kc + bk) * Np + n0 + bn);
      *reinterpret_cast<uint4*>(&smem[bBase + bk * 64 + bn]) = bv;
    }
#endif
    if (has_next && (kc + 64) < Kp) {  // gfx1250 global_prefetch_b8, two chunks ahead
      __builtin_prefetch(A + (size_t)(m0 + (t >> 1)) * Kp + kc + 64 + ((t & 1) << 4), 0, 3);
      __builtin_prefetch(B + (size_t)(kc + 64 + (t >> 3)) * Np + n0 + ((t & 7) << 3), 0, 3);
    }
    __syncthreads();

    // Assemble fragments per CDNA5 16-bit WMMA VGPR layouts (wave32).
    v16h afrag[2], bfrag[2];
#pragma unroll
    for (int i = 0; i < 2; ++i) {
      const int row = wy * 32 + i * 16 + lo;       // A: lane lo -> M row
#pragma unroll
      for (int v = 0; v < 8; ++v) {
        // K pair held by VGPR v: (v>=4?16:0) + hi*8 + (v&3)*2
        const int kk = ((v & 4) << 2) + (hi << 3) + ((v & 3) << 1);
        afrag[i][2 * v]     = smem[aBase + row * 32 + kk];
        afrag[i][2 * v + 1] = smem[aBase + row * 32 + kk + 1];
      }
    }
#pragma unroll
    for (int j = 0; j < 2; ++j) {
      const int col = wx * 32 + j * 16 + lo;       // B: lane lo -> N col
#pragma unroll
      for (int v = 0; v < 8; ++v) {
        const int kk = ((v & 4) << 2) + (hi << 3) + ((v & 3) << 1);       // K pair: (v>=4?16:0) + hi*8 + (v&3)*2
        bfrag[j][2 * v]     = smem[bBase + kk * 64 + col];
        bfrag[j][2 * v + 1] = smem[bBase + (kk + 1) * 64 + col];
      }
    }

#pragma unroll
    for (int i = 0; i < 2; ++i)
#pragma unroll
      for (int j = 0; j < 2; ++j)
        acc[i][j] = __builtin_amdgcn_wmma_f32_16x16x32_f16(
            false, afrag[i], false, bfrag[j], (short)0, acc[i][j], false, false);

    __syncthreads();
    buf ^= 1;
  }

  // Epilogue: C layout — VGPR v holds M = v + 8*hi, N = lo
#pragma unroll
  for (int i = 0; i < 2; ++i) {
#pragma unroll
    for (int j = 0; j < 2; ++j) {
      const int col = n0 + wx * 32 + j * 16 + lo;
#pragma unroll
      for (int v = 0; v < 8; ++v) {
        const int row = m0 + wy * 32 + i * 16 + hi * 8 + v;
        C[(size_t)row * Np + col] = acc[i][j][v];
      }
    }
  }
}

// ---------------- attention coefficients: als/ald [N,heads] ----------------
__global__ void k_att(const float* __restrict__ H, const float* __restrict__ a_src,
                      const float* __restrict__ a_dst, float* __restrict__ als,
                      float* __restrict__ ald, int N, int heads, int ch, int stride) {
  int i = blockIdx.x * blockDim.x + threadIdx.x;
  if (i >= N * heads) return;
  int n = i / heads, h = i % heads;
  const float* row = H + (size_t)n * stride + h * ch;
  const float* as  = a_src + h * ch;
  const float* ad  = a_dst + h * ch;
  float s = 0.f, d = 0.f;
  for (int c = 0; c < ch; ++c) { float v = row[c]; s += v * as[c]; d += v * ad[c]; }
  als[i] = s;
  ald[i] = d;
}

// ---------------- float atomic max (int/uint trick, init -inf safe) ----------------
__device__ inline void atomicMaxF(float* addr, float val) {
  if (val >= 0.f) atomicMax(reinterpret_cast<int*>(addr), __float_as_int(val));
  else            atomicMin(reinterpret_cast<unsigned int*>(addr), __float_as_uint(val));
}

__device__ inline void edge_sd(const int* __restrict__ ei, int E, int e, int& s, int& d) {
  if (e < E) { s = ei[e]; d = ei[E + e]; }
  else       { s = d = e - E; }          // appended self-loops
}

// ---------------- edge pass 1: segment max of leaky_relu(als[s]+ald[d]) ----------------
__global__ void k_edge_max(const int* __restrict__ ei, int E, int Etot,
                           const float* __restrict__ als, const float* __restrict__ ald,
                           float* m, int heads) {
  int e = blockIdx.x * blockDim.x + threadIdx.x;
  if (e >= Etot) return;
  int s, d; edge_sd(ei, E, e, s, d);
  for (int h = 0; h < heads; ++h) {
    float v = als[s * heads + h] + ald[d * heads + h];
    v = v >= 0.f ? v : NEG_SLOPE * v;
    atomicMaxF(&m[d * heads + h], v);
  }
}

// ---------------- edge pass 2: p = exp(e - m[dst]); segment sum ----------------
__global__ void k_edge_sum(const int* __restrict__ ei, int E, int Etot,
                           const float* __restrict__ als, const float* __restrict__ ald,
                           const float* __restrict__ m, float* ssum,
                           float* __restrict__ pbuf, int heads) {
  int e = blockIdx.x * blockDim.x + threadIdx.x;
  if (e >= Etot) return;
  int s, d; edge_sd(ei, E, e, s, d);
  for (int h = 0; h < heads; ++h) {
    float v = als[s * heads + h] + ald[d * heads + h];
    v = v >= 0.f ? v : NEG_SLOPE * v;
    float p = expf(v - m[d * heads + h]);
    pbuf[(size_t)e * heads + h] = p;
    atomicAdd(&ssum[d * heads + h], p);
  }
}

// ---------------- edge pass 3: out[dst] += p * H[src] ----------------
__global__ void k_edge_aggr(const int* __restrict__ ei, int E,
                            const float* __restrict__ pbuf, const float* __restrict__ H,
                            float* out, int heads, int ch, int stride) {
  int e = blockIdx.x;
  int s, d; edge_sd(ei, E, e, s, d);
  const int F = heads * ch;
  for (int col = threadIdx.x; col < F; col += blockDim.x) {
    float p = pbuf[(size_t)e * heads + col / ch];
    atomicAdd(&out[(size_t)d * stride + col], p * H[(size_t)s * stride + col]);
  }
}

// ---------------- layer-1 epilogue: normalize + bias + ELU -> padded f16 ----------------
__global__ void k_finish1(const float* __restrict__ out1, const float* __restrict__ s1,
                          const float* __restrict__ b1, _Float16* __restrict__ X2h,
                          int N, int Mp) {
  long i = (long)blockIdx.x * blockDim.x + threadIdx.x;
  long total = (long)Mp * 768;
  if (i >= total) return;
  int n = (int)(i / 768), col = (int)(i % 768);
  float v = 0.f;
  if (n < N && col < 750) {
    int h = col / 75;
    v = out1[i] / (s1[n * 10 + h] + 1e-16f) + b1[col];
    v = v > 0.f ? v : expm1f(v);   // ELU
  }
  X2h[i] = (_Float16)v;
}

// ---------------- layer-2 epilogue + fused graph max pool ----------------
__global__ void k_finish2_pool(const float* __restrict__ out2, const float* __restrict__ s2,
                               const float* __restrict__ b2, const int* __restrict__ batch,
                               float* g, int N) {
  int i = blockIdx.x * blockDim.x + threadIdx.x;
  if (i >= N * 128) return;
  int n = i / 128, c = i % 128;
  float v = out2[(size_t)n * 128 + c] / (s2[n] + 1e-16f) + b2[c];
  v = fmaxf(v, 0.f);                         // ReLU
  atomicMaxF(&g[batch[n] * 128 + c], v);     // global max pool per graph
}

// ---------------- head: relu(g_clean @ Wg + bg), 64x128 ----------------
__global__ void k_head(const float* __restrict__ g, const float* __restrict__ Wg,
                       const float* __restrict__ bg, float* __restrict__ out) {
  int i = blockIdx.x * blockDim.x + threadIdx.x;
  if (i >= NGRAPHS * 128) return;
  int r = i / 128, c = i % 128;
  float acc = bg[c];
  for (int k = 0; k < 128; ++k) {
    float gv = fmaxf(g[r * 128 + k], 0.f);   // empty graph (-inf) -> 0; pooled vals >= 0
    acc += gv * Wg[k * 128 + c];
  }
  out[i] = fmaxf(acc, 0.f);
}

extern "C" void kernel_launch(void* const* d_in, const int* in_sizes, int n_in,
                              void* d_out, int out_size, void* d_ws, size_t ws_size,
                              hipStream_t stream) {
  const float* x    = (const float*)d_in[0];
  const int*   ei   = (const int*)d_in[1];
  const int*   batch= (const int*)d_in[2];
  const float* W1   = (const float*)d_in[3];
  const float* a1s  = (const float*)d_in[4];
  const float* a1d  = (const float*)d_in[5];
  const float* b1   = (const float*)d_in[6];
  const float* W2   = (const float*)d_in[7];
  const float* a2s  = (const float*)d_in[8];
  const float* a2d  = (const float*)d_in[9];
  const float* b2   = (const float*)d_in[10];
  const float* Wg   = (const float*)d_in[11];
  const float* bg   = (const float*)d_in[12];
  float* out = (float*)d_out;

  const int N    = in_sizes[0] / 75;         // 10000
  const int E    = in_sizes[1] / 2;          // 320000
  const int Etot = E + N;                    // + self loops
  const int Mp   = ((N + 127) / 128) * 128;  // 10112

  // ---- carve workspace ----
  uint8_t* base = (uint8_t*)d_ws;
  size_t off = 0;
  auto carve = [&](size_t bytes) -> void* {
    void* p = base + off;
    off = (off + bytes + 255) & ~(size_t)255;
    return p;
  };
  _Float16* xh   = (_Float16*)carve((size_t)Mp * 96 * 2);
  _Float16* w1h  = (_Float16*)carve((size_t)96 * 768 * 2);
  _Float16* w2h  = (_Float16*)carve((size_t)768 * 128 * 2);
  float* H1      = (float*)carve((size_t)Mp * 768 * 4);
  float* out1    = (float*)carve((size_t)Mp * 768 * 4);
  _Float16* X2h  = (_Float16*)carve((size_t)Mp * 768 * 2);
  float* H2      = (float*)carve((size_t)Mp * 128 * 4);
  float* out2    = (float*)carve((size_t)Mp * 128 * 4);
  float* als1    = (float*)carve((size_t)N * 10 * 4);
  float* ald1    = (float*)carve((size_t)N * 10 * 4);
  float* m1      = (float*)carve((size_t)N * 10 * 4);
  float* s1      = (float*)carve((size_t)N * 10 * 4);
  float* pbuf1   = (float*)carve((size_t)Etot * 10 * 4);
  float* als2    = (float*)carve((size_t)N * 4);
  float* ald2    = (float*)carve((size_t)N * 4);
  float* m2      = (float*)carve((size_t)N * 4);
  float* s2      = (float*)carve((size_t)N * 4);
  float* pbuf2   = (float*)carve((size_t)Etot * 4);
  float* g       = (float*)carve((size_t)NGRAPHS * 128 * 4);

  const int T = 256;
  // ---- pack to padded f16 ----
  k_pack_h<<<ceil_div((long)Mp * 96, T), T, 0, stream>>>(x, xh, N, 75, Mp, 96);
  k_pack_h<<<ceil_div((long)96 * 768, T), T, 0, stream>>>(W1, w1h, 75, 750, 96, 768);
  k_pack_h<<<ceil_div((long)768 * 128, T), T, 0, stream>>>(W2, w2h, 750, 128, 768, 128);

  // ---- init accumulators ----
  k_fill_f32<<<ceil_div((long)N * 10, T), T, 0, stream>>>(m1, (long)N * 10, -INFINITY);
  k_fill_f32<<<ceil_div((long)N * 10, T), T, 0, stream>>>(s1, (long)N * 10, 0.f);
  k_fill_f32<<<ceil_div((long)Mp * 768, T), T, 0, stream>>>(out1, (long)Mp * 768, 0.f);
  k_fill_f32<<<ceil_div((long)N, T), T, 0, stream>>>(m2, (long)N, -INFINITY);
  k_fill_f32<<<ceil_div((long)N, T), T, 0, stream>>>(s2, (long)N, 0.f);
  k_fill_f32<<<ceil_div((long)Mp * 128, T), T, 0, stream>>>(out2, (long)Mp * 128, 0.f);
  k_fill_f32<<<ceil_div((long)NGRAPHS * 128, T), T, 0, stream>>>(g, (long)NGRAPHS * 128, -INFINITY);

  // ---- layer 1: H1 = x @ W1 (TDM -> LDS -> WMMA) ----
  k_gemm_wmma<<<dim3(768 / 64, Mp / 128), 256, GEMM_LDS_BYTES, stream>>>(xh, w1h, H1, Mp, 768, 96);
  k_att<<<ceil_div((long)N * 10, T), T, 0, stream>>>(H1, a1s, a1d, als1, ald1, N, 10, 75, 768);
  k_edge_max<<<ceil_div((long)Etot, T), T, 0, stream>>>(ei, E, Etot, als1, ald1, m1, 10);
  k_edge_sum<<<ceil_div((long)Etot, T), T, 0, stream>>>(ei, E, Etot, als1, ald1, m1, s1, pbuf1, 10);
  k_edge_aggr<<<Etot, 256, 0, stream>>>(ei, E, pbuf1, H1, out1, 10, 75, 768);
  k_finish1<<<ceil_div((long)Mp * 768, T), T, 0, stream>>>(out1, s1, b1, X2h, N, Mp);

  // ---- layer 2: H2 = elu_out @ W2 (TDM -> LDS -> WMMA) ----
  k_gemm_wmma<<<dim3(128 / 64, Mp / 128), 256, GEMM_LDS_BYTES, stream>>>(X2h, w2h, H2, Mp, 128, 768);
  k_att<<<ceil_div((long)N, T), T, 0, stream>>>(H2, a2s, a2d, als2, ald2, N, 1, 128, 128);
  k_edge_max<<<ceil_div((long)Etot, T), T, 0, stream>>>(ei, E, Etot, als2, ald2, m2, 1);
  k_edge_sum<<<ceil_div((long)Etot, T), T, 0, stream>>>(ei, E, Etot, als2, ald2, m2, s2, pbuf2, 1);
  k_edge_aggr<<<Etot, 128, 0, stream>>>(ei, E, pbuf2, H2, out2, 1, 128, 128);
  k_finish2_pool<<<ceil_div((long)N * 128, T), T, 0, stream>>>(out2, s2, b2, batch, g, N);

  // ---- head ----
  k_head<<<ceil_div((long)NGRAPHS * 128, T), T, 0, stream>>>(g, Wg, bg, out);
}
